// TRED_GNN_82437602279986
// MI455X (gfx1250) — matrix operand order repaired and checked by reading, not compiled, not run on hardware
//
#include <hip/hip_runtime.h>
#include <math.h>

typedef __attribute__((ext_vector_type(16))) _Float16 v16h;
typedef __attribute__((ext_vector_type(8)))  float    v8f;

// ---------------------------------------------------------------------------
// Zero-fill (agg accumulator must be zeroed every call; harness doesn't)
// ---------------------------------------------------------------------------
__global__ void zero_f32(float* __restrict__ p, long n) {
  long i = (long)blockIdx.x * blockDim.x + threadIdx.x;
  long stride = (long)gridDim.x * blockDim.x;
  for (; i < n; i += stride) p[i] = 0.0f;
}

// ---------------------------------------------------------------------------
// Convert + transpose a 64x64 f32 weight W[k][n] into f16 WT[n][k].
// After this, WMMA B-fragments are 16 contiguous halves per lane.
// ---------------------------------------------------------------------------
__global__ __launch_bounds__(256)
void cvt_transpose_w(const float* __restrict__ W, _Float16* __restrict__ WT) {
  for (int i = threadIdx.x; i < 64 * 64; i += 256) {
    const int k = i >> 6;
    const int n = i & 63;
    WT[n * 64 + k] = (_Float16)W[i];
  }
}

// ---------------------------------------------------------------------------
// Y[M,64] = (relu?)( X[M,64] @ W[64,64] ), W pre-transposed to f16 (WT[n][k]).
// One wave owns a 16-row stripe: 4 n-tiles x (K=64 as 2 steps of 32)
// = 8 x v_wmma_f32_16x16x32_f16. blockDim = 128 (4 waves).
// GUARD=false (M % 16 == 0): zero boundary code in the instruction stream.
// GUARD=true: clamp addresses + whole-fragment zero-select (no per-element
// exec-mask branches). X and Y may alias (in-place): a row is only
// read/written by its own wave, and all reads precede all stores.
// ---------------------------------------------------------------------------
template<bool RELU, bool GUARD>
__global__ __launch_bounds__(128)
void gemm64_wmma(const float* X, const _Float16* __restrict__ WT, float* Y, int M)
{
  const int lane = threadIdx.x & 31;
  const int wv   = threadIdx.x >> 5;
  const int row0 = (blockIdx.x * 4 + wv) * 16;
  if (row0 >= M) return;                    // whole-wave exit; EXEC stays all-1s
  const int g  = lane >> 4;                 // half-wave group
  const int ln = lane & 15;

  // ---- B fragments: contiguous k order (matches A): lanes 0-15 hold K=k0..k0+15, lanes 16-31
  //      K=k0+16..k0+31; col = ln. WT[n][k] is contiguous in k -> v16h load.
  v16h bfrag[2][4];
  #pragma unroll
  for (int ks = 0; ks < 2; ++ks) {
    #pragma unroll
    for (int t = 0; t < 4; ++t) {
      bfrag[ks][t] = *(const v16h*)(WT + (t * 16 + ln) * 64 + ks * 32 + 16 * g);
    }
  }

  // ---- A fragments: row = row0 + ln; element j <-> K =
  //      k0 + 16*g + j  => one contiguous 16-float run per step.
  const int rowm = row0 + ln;
  const int rowc = GUARD ? (rowm < M ? rowm : (M - 1)) : rowm;
  const float4* xrow4 = (const float4*)(X + (size_t)rowc * 64);

  v16h afrag[2];
  #pragma unroll
  for (int ks = 0; ks < 2; ++ks) {
    const int base = (ks * 32 + 16 * g) >> 2;        // in float4 units
    const float4 r0 = xrow4[base];
    const float4 r1 = xrow4[base + 1];
    const float4 r2 = xrow4[base + 2];               // +8 floats
    const float4 r3 = xrow4[base + 3];
    v16h a;
    a[0]  = (_Float16)r0.x; a[1]  = (_Float16)r0.y;
    a[2]  = (_Float16)r0.z; a[3]  = (_Float16)r0.w;
    a[4]  = (_Float16)r1.x; a[5]  = (_Float16)r1.y;
    a[6]  = (_Float16)r1.z; a[7]  = (_Float16)r1.w;
    a[8]  = (_Float16)r2.x; a[9]  = (_Float16)r2.y;
    a[10] = (_Float16)r2.z; a[11] = (_Float16)r2.w;
    a[12] = (_Float16)r3.x; a[13] = (_Float16)r3.y;
    a[14] = (_Float16)r3.z; a[15] = (_Float16)r3.w;
    afrag[ks] = a;
  }
  if (GUARD) {                                       // tail tiles only
    const v16h z = {};
    if (rowm >= M) { afrag[0] = z; afrag[1] = z; }   // data select, not branches
  }

  // ---- accumulate
  v8f acc[4] = {};
  #pragma unroll
  for (int ks = 0; ks < 2; ++ks) {
    #pragma unroll
    for (int t = 0; t < 4; ++t) {
      acc[t] = __builtin_amdgcn_wmma_f32_16x16x32_f16(
          /*neg_a=*/false, afrag[ks], /*neg_b=*/false, bfrag[ks][t],
          /*c_mod=*/(short)0, acc[t], /*reuse_a=*/false, /*reuse_b=*/false);
    }
  }

  // ---- D layout: lane = (m>=8?16:0)+n, vgpr r = m&7 -> m = row0 + 8*g + r
  #pragma unroll
  for (int t = 0; t < 4; ++t) {
    #pragma unroll
    for (int r = 0; r < 8; ++r) {
      const int m = row0 + 8 * g + r;
      float v = acc[t][r];
      if (RELU) v = fmaxf(v, 0.0f);
      if (!GUARD || m < M) Y[(size_t)m * 64 + t * 16 + ln] = v;
    }
  }
}

// ---------------------------------------------------------------------------
// Edge kernel: one wave per edge, 2 dims (float2) per lane.
// pre = HWs[sub] + RWr[rel] + RWqr[q_rel[r_idx]] + bqr ; relu ; dot Wa ; +ba
// alpha = sigmoid(.) ; msg = alpha*(hidden[sub]+rela[rel]) ; atomicAdd agg[obj]
// ---------------------------------------------------------------------------
__global__ __launch_bounds__(256)
void edge_attn_scatter(const float* __restrict__ hidden,
                       const float* __restrict__ HWs,
                       const float* __restrict__ RWr,
                       const float* __restrict__ RWqr,
                       const float* __restrict__ rela,
                       const int*   __restrict__ q_rel,
                       const int*   __restrict__ r_idx,
                       const int*   __restrict__ rel,
                       const int*   __restrict__ sub,
                       const int*   __restrict__ obj,
                       const float* __restrict__ bqr,
                       const float* __restrict__ Wa,
                       const float* __restrict__ ba,
                       float*       __restrict__ agg,
                       int NE)
{
  const int e    = blockIdx.x * (blockDim.x >> 5) + (threadIdx.x >> 5);
  const int lane = threadIdx.x & 31;
  if (e >= NE) return;

  const int s  = sub[e];
  const int o  = obj[e];
  const int rl = rel[e];
  const int q  = q_rel[r_idx[e]];

  const float2* hs2 = (const float2*)(hidden + (size_t)s  * 64);
  const float2* hw2 = (const float2*)(HWs    + (size_t)s  * 64);
  const float2* rw2 = (const float2*)(RWr    + (size_t)rl * 64);
  const float2* qw2 = (const float2*)(RWqr   + (size_t)q  * 64);
  const float2* hr2 = (const float2*)(rela   + (size_t)rl * 64);
  const float2* bq2 = (const float2*)bqr;
  const float2* wa2 = (const float2*)Wa;

  const float2 hw = hw2[lane];
  const float2 rw = rw2[lane];
  const float2 qw = qw2[lane];
  const float2 bq = bq2[lane];
  const float2 wa = wa2[lane];

  float px = fmaxf(hw.x + rw.x + qw.x + bq.x, 0.0f);
  float py = fmaxf(hw.y + rw.y + qw.y + bq.y, 0.0f);
  float part = px * wa.x + py * wa.y;

  #pragma unroll
  for (int off = 16; off >= 1; off >>= 1)
    part += __shfl_xor(part, off, 32);

  const float alpha = 1.0f / (1.0f + __expf(-(part + ba[0])));

  const float2 hs = hs2[lane];
  const float2 hr = hr2[lane];
  float* dst = agg + (size_t)o * 64 + lane * 2;
  atomicAdd(dst,     alpha * (hs.x + hr.x));
  atomicAdd(dst + 1, alpha * (hs.y + hr.y));
}

// ---------------------------------------------------------------------------
extern "C" void kernel_launch(void* const* d_in, const int* in_sizes, int n_in,
                              void* d_out, int out_size, void* d_ws, size_t ws_size,
                              hipStream_t stream)
{
  (void)n_in; (void)ws_size; (void)out_size;

  const int*   q_rel  = (const int*)  d_in[0];
  const float* hidden = (const float*)d_in[1];
  const int*   r_idx  = (const int*)  d_in[2];
  const int*   rel    = (const int*)  d_in[3];
  const int*   sub    = (const int*)  d_in[4];
  const int*   obj    = (const int*)  d_in[5];
  /* d_in[6] = n_node device scalar; derive from in_sizes instead */
  const float* rela   = (const float*)d_in[7];
  const float* Ws     = (const float*)d_in[8];
  const float* Wr     = (const float*)d_in[9];
  const float* Wqr    = (const float*)d_in[10];
  const float* bqr    = (const float*)d_in[11];
  const float* Wa     = (const float*)d_in[12];
  const float* ba     = (const float*)d_in[13];
  const float* Wh     = (const float*)d_in[14];

  const int NN    = in_sizes[1] / 64;   // 200000
  const int NE    = in_sizes[2];        // 1200000
  const int NRELS = in_sizes[7] / 64;   // 461

  // Workspace layout (256B-aligned slots):
  //   HWs [NN,64] f32 | RWr [NRELS,64] f32 | RWqr [NRELS,64] f32
  //   | 4x WT (64x64 f16, transposed)
  char*  ws = (char*)d_ws;
  size_t off = 0;
  float* HWs  = (float*)(ws + off); off += (size_t)NN * 64 * sizeof(float);
  off = (off + 255) & ~(size_t)255;
  float* RWr  = (float*)(ws + off); off += (size_t)((NRELS + 15) & ~15) * 64 * sizeof(float);
  off = (off + 255) & ~(size_t)255;
  float* RWqr = (float*)(ws + off); off += (size_t)((NRELS + 15) & ~15) * 64 * sizeof(float);
  off = (off + 255) & ~(size_t)255;
  _Float16* WTs  = (_Float16*)(ws + off); off += 64 * 64 * sizeof(_Float16);
  _Float16* WTr  = (_Float16*)(ws + off); off += 64 * 64 * sizeof(_Float16);
  _Float16* WTqr = (_Float16*)(ws + off); off += 64 * 64 * sizeof(_Float16);
  _Float16* WTh  = (_Float16*)(ws + off); off += 64 * 64 * sizeof(_Float16);

  float* agg = (float*)d_out;           // accumulate directly in d_out

  // Weight convert+transpose (tiny)
  cvt_transpose_w<<<1, 256, 0, stream>>>(Ws,  WTs);
  cvt_transpose_w<<<1, 256, 0, stream>>>(Wr,  WTr);
  cvt_transpose_w<<<1, 256, 0, stream>>>(Wqr, WTqr);
  cvt_transpose_w<<<1, 256, 0, stream>>>(Wh,  WTh);

  // Zero the aggregation buffer
  zero_f32<<<1024, 256, 0, stream>>>(agg, (long)NN * 64);

  // Dense precomputes via WMMA (guard-free instantiation when M % 16 == 0)
  if ((NN & 15) == 0)
    gemm64_wmma<false, false><<<(NN + 63) / 64, 128, 0, stream>>>(hidden, WTs, HWs, NN);
  else
    gemm64_wmma<false, true ><<<(NN + 63) / 64, 128, 0, stream>>>(hidden, WTs, HWs, NN);

  gemm64_wmma<false, true><<<(NRELS + 63) / 64, 128, 0, stream>>>(rela, WTr,  RWr,  NRELS);
  gemm64_wmma<false, true><<<(NRELS + 63) / 64, 128, 0, stream>>>(rela, WTqr, RWqr, NRELS);

  // Edge gather / attention / scatter-add (memory-bound phase)
  const int edges_per_block = 256 / 32;
  edge_attn_scatter<<<(NE + edges_per_block - 1) / edges_per_block, 256, 0, stream>>>(
      hidden, HWs, RWr, RWqr, rela, q_rel, r_idx, rel, sub, obj,
      bqr, Wa, ba, agg, NE);

  // Final in-place: d_out = relu(agg @ Wh)
  if ((NN & 15) == 0)
    gemm64_wmma<true, false><<<(NN + 63) / 64, 128, 0, stream>>>(agg, WTh, (float*)d_out, NN);
  else
    gemm64_wmma<true, true ><<<(NN + 63) / 64, 128, 0, stream>>>(agg, WTh, (float*)d_out, NN);
}
